// AccumulatingFeedForwardKernel_85615878078677
// MI455X (gfx1250) — matrix operand order repaired and checked by reading, not compiled, not run on hardware
//
#include <hip/hip_runtime.h>
#include <cstdint>

// Problem constants (from reference): B=16, K=4096, Q=64, H=256
#define B_    16
#define K_    4096
#define Q_    64
#define H_    256
#define NWG_  32      // workgroups cooperating per batch
#define KPW_  128     // keys per workgroup (8 waves x 16-key WMMA tiles)
#define STRH_ 264     // padded f16 LDS row stride in halves: bank skew 4m, keeps 16B align

typedef __attribute__((ext_vector_type(4)))  float     f4;
typedef __attribute__((ext_vector_type(4)))  _Float16  v4h;
typedef __attribute__((ext_vector_type(8)))  _Float16  v8h;
typedef __attribute__((ext_vector_type(16))) _Float16  v16h;
typedef __attribute__((ext_vector_type(8)))  float     v8f;
typedef __attribute__((__vector_size__(16))) int       i4v;   // matches builtin param

// ---- hardware tanh (gfx1250 has V_TANH_F32) with safe fallback ----
__device__ __forceinline__ float fast_tanh_fb(float x) {
  float ax = fabsf(x);
  float t  = __expf(-2.0f * ax);
  float r  = 1.0f - 2.0f * t / (1.0f + t);
  return copysignf(r, x);
}
#if __has_builtin(__builtin_amdgcn_tanhf)
#define TANH_FAST(x) __builtin_amdgcn_tanhf(x)
#elif __has_builtin(__builtin_amdgcn_tanh_f32)
#define TANH_FAST(x) __builtin_amdgcn_tanh_f32(x)
#else
#define TANH_FAST(x) fast_tanh_fb(x)
#endif

// ---- async global->LDS copy (CDNA5 GLOBAL_LOAD_ASYNC_TO_LDS_B128) ----
// Signature (from compiler diagnostic): arg0 = int4 ptr in AS1 (__device__),
// arg1 = int4 ptr in AS3 (LDS), then imm offset, imm cpol.
__device__ __forceinline__ void async_copy_b128(const float* g, float* lds) {
#if __has_builtin(__builtin_amdgcn_global_load_async_to_lds_b128)
  __builtin_amdgcn_global_load_async_to_lds_b128(
      (__attribute__((address_space(1))) i4v*)(uintptr_t)g,
      (__attribute__((address_space(3))) i4v*)(uint32_t)(uintptr_t)lds,
      0, 0);
#else
  *(f4*)lds = *(const f4*)g;
#endif
}

__device__ __forceinline__ void wait_async_all() {
#if __has_builtin(__builtin_amdgcn_s_wait_asynccnt)
  __builtin_amdgcn_s_wait_asynccnt(0);
#elif defined(__AMDGCN__)
  asm volatile("s_wait_asynccnt 0" ::: "memory");
#endif
}

__device__ __forceinline__ void spin_pause() {
#if __has_builtin(__builtin_amdgcn_s_sleep)
  __builtin_amdgcn_s_sleep(1);
#endif
}

// zero the per-(b,step) arrival counters each call (workspace persists across replays)
__global__ void affw_init(unsigned* __restrict__ gcnt) {
  int t = blockIdx.x * blockDim.x + threadIdx.x;
  if (t < B_ * Q_) gcnt[t] = 0u;
}

__global__ __launch_bounds__(256) void affw_main(
    const float* __restrict__ key,          // (B,K,H)
    const float* __restrict__ query,        // (Q,B,H)
    const unsigned char* __restrict__ mask, // (B,K) bool
    const float* __restrict__ wvec,         // (H)
    const float* __restrict__ awvec,        // (H)
    float* __restrict__ out,                // attention (B,Q,K) then acc (B,1,K)
    float* __restrict__ gpart,              // (B*Q, NWG) partial exp-sums
    unsigned* __restrict__ gcnt)            // (B*Q) arrival counters
{
  __shared__ __align__(16) _Float16 keyTile[KPW_ * STRH_]; // 66 KB f16 key tile
  __shared__ __align__(16) float    qbuf[2][H_];           // double-buffered query row
  __shared__ __align__(16) float    awbuf[H_];
  __shared__ __align__(32) _Float16 wbuf[H_];
  __shared__ float sLDS[KPW_];
  __shared__ float accLDS[KPW_];
  __shared__ float wsum[8];
  __shared__ float denomLDS;

  const int tid  = threadIdx.x;
  const int wg   = blockIdx.x;   // 0..NWG_-1 : K-chunk within batch
  const int b    = blockIdx.y;   // 0..B_-1
  const int kbase = wg * KPW_;
  const int lane = tid & 31;
  const int wid  = tid >> 5;     // wave 0..7 -> keys [wid*16, wid*16+16)
  const int m16  = lane & 15;    // WMMA row M
  const int g    = lane >> 4;    // lane group (ISA A/B layout K offset)

  // one-time parameter loads
  if (tid < H_) { awbuf[tid] = awvec[tid]; wbuf[tid] = (_Float16)wvec[tid]; }
  if (tid < KPW_) accLDS[tid] = 0.0f;

  // one-time key tile load: f32 global -> f16 LDS (padded rows)
  {
    const float* kg = key + ((size_t)b * K_ + kbase) * H_;
    #pragma unroll
    for (int it = 0; it < (KPW_ * H_) / (256 * 4); ++it) {
      int i4 = tid + it * 256;
      int m  = i4 >> 6;              // row (64 float4 per row)
      int hq = (i4 & 63) << 2;       // h offset
      f4 v = *(const f4*)(kg + (size_t)m * H_ + hq);
      v4h h; h[0] = (_Float16)v[0]; h[1] = (_Float16)v[1];
             h[2] = (_Float16)v[2]; h[3] = (_Float16)v[3];
      *(v4h*)&keyTile[m * STRH_ + hq] = h;
    }
  }

  // async prefetch of q row for step 0
  if (tid < 64) async_copy_b128(query + (size_t)b * H_ + tid * 4, &qbuf[0][tid * 4]);
  wait_async_all();
  __syncthreads();

  for (int s = 0; s < Q_; ++s) {
    const float* qb = qbuf[s & 1];
    // async prefetch next step's query row into the other buffer
    if (s + 1 < Q_ && tid < 64)
      async_copy_b128(query + ((size_t)(s + 1) * B_ + b) * H_ + tid * 4,
                      &qbuf[(s + 1) & 1][tid * 4]);

    // ---- scores via v_wmma_f32_16x16x32_f16, accumulating over 8 h-chunks ----
    const float accv = accLDS[wid * 16 + m16];
    const int   row  = (wid * 16 + m16) * STRH_;
    v8f cacc = {};
    #pragma unroll
    for (int c = 0; c < 8; ++c) {
      const int h0 = c * 32 + g * 8;   // A layout: j<8  -> K = j + 8g
      const int h1 = h0 + 16;          //           j>=8 -> K = (j-8) + 16 + 8g
      v8h k0 = *(const v8h*)&keyTile[row + h0];
      v8h k1 = *(const v8h*)&keyTile[row + h1];
      f4 q0a = *(const f4*)&qb[h0];     f4 q0b = *(const f4*)&qb[h0 + 4];
      f4 q1a = *(const f4*)&qb[h1];     f4 q1b = *(const f4*)&qb[h1 + 4];
      f4 a0a = *(const f4*)&awbuf[h0];  f4 a0b = *(const f4*)&awbuf[h0 + 4];
      f4 a1a = *(const f4*)&awbuf[h1];  f4 a1b = *(const f4*)&awbuf[h1 + 4];
      union { v16h v; _Float16 e[16]; } a;
      #pragma unroll
      for (int j = 0; j < 4; ++j) {
        a.e[j]      = (_Float16)TANH_FAST((float)k0[j]     + q0a[j] - a0a[j] * accv);
        a.e[j + 4]  = (_Float16)TANH_FAST((float)k0[j + 4] + q0b[j] - a0b[j] * accv);
        a.e[j + 8]  = (_Float16)TANH_FAST((float)k1[j]     + q1a[j] - a1a[j] * accv);
        a.e[j + 12] = (_Float16)TANH_FAST((float)k1[j + 4] + q1b[j] - a1b[j] * accv);
      }
      // B layout (32x16 f16): same K order as A (j<8 -> K=j+8g, j>=8 -> K=(j-8)+16+8g);
      // all 16 columns identical = weight chunk -> D columns all equal scores.
      v8h b0 = *(const v8h*)&wbuf[h0];
      v8h b1 = *(const v8h*)&wbuf[h1];
      v16h bm = __builtin_shufflevector(b0, b1, 0,1,2,3,4,5,6,7,8,9,10,11,12,13,14,15);
      cacc = __builtin_amdgcn_wmma_f32_16x16x32_f16(
                 false, a.v, false, bm, (short)0, cacc, false, false);
    }
    // C layout: lanes 0-15 hold rows 0-7, lanes 16-31 rows 8-15 (any column)
    if (m16 == 0) {
      const int basek = wid * 16 + g * 8;
      #pragma unroll
      for (int r = 0; r < 8; ++r) sLDS[basek + r] = cacc[r];
    }
    __syncthreads(); // (A) scores visible

    // ---- masked exp + cooperative softmax denominator ----
    float e = 0.0f;
    if (tid < KPW_) {
      float sc = sLDS[tid];
      bool  mk = mask[(size_t)b * K_ + kbase + tid] != 0;
      e = mk ? 0.0f : expf(sc); // |score| small enough: skip max-subtraction
    }
    float es = e;
    #pragma unroll
    for (int off = 16; off > 0; off >>= 1) es += __shfl_down(es, off, 32);
    if (lane == 0) wsum[wid] = es;
    __syncthreads(); // (B)

    if (tid == 0) {
      float p = 0.0f;
      #pragma unroll
      for (int i = 0; i < 8; ++i) p += wsum[i];
      const int slot = b * Q_ + s;
      __hip_atomic_store(&gpart[(size_t)slot * NWG_ + wg], p,
                         __ATOMIC_RELAXED, __HIP_MEMORY_SCOPE_AGENT);
      __hip_atomic_fetch_add(&gcnt[slot], 1u,
                             __ATOMIC_RELEASE, __HIP_MEMORY_SCOPE_AGENT);
      while (__hip_atomic_load(&gcnt[slot], __ATOMIC_ACQUIRE,
                               __HIP_MEMORY_SCOPE_AGENT) < (unsigned)NWG_)
        spin_pause();
      float d = 0.0f;
      for (int i = 0; i < NWG_; ++i)  // fixed order -> deterministic
        d += __hip_atomic_load(&gpart[(size_t)slot * NWG_ + i],
                               __ATOMIC_RELAXED, __HIP_MEMORY_SCOPE_AGENT);
      denomLDS = d;
    }
    __syncthreads(); // (C) denominator visible

    const float inv = 1.0f / denomLDS;
    if (tid < KPW_) {
      float att = e * inv;
      __builtin_nontemporal_store(att,
          &out[((size_t)b * Q_ + s) * K_ + kbase + tid]); // streamed, never re-read
      accLDS[tid] += att;
    }
    wait_async_all();  // next q row landed in LDS
    __syncthreads();   // (D) acc update + q buffer visible for next step
  }

  if (tid < KPW_)
    out[(size_t)B_ * Q_ * K_ + (size_t)b * K_ + kbase + tid] = accLDS[tid];
}

extern "C" void kernel_launch(void* const* d_in, const int* in_sizes, int n_in,
                              void* d_out, int out_size, void* d_ws, size_t ws_size,
                              hipStream_t stream) {
  (void)in_sizes; (void)n_in; (void)out_size; (void)ws_size;
  const float*         key   = (const float*)d_in[0];
  const float*         query = (const float*)d_in[1];
  const unsigned char* mask  = (const unsigned char*)d_in[2]; // numpy bool, 1B
  const float*         wv    = (const float*)d_in[3];
  const float*         awv   = (const float*)d_in[4];
  float*    out   = (float*)d_out;
  float*    gpart = (float*)d_ws;                       // B*Q*NWG floats
  unsigned* gcnt  = (unsigned*)(gpart + (size_t)B_ * Q_ * NWG_);

  hipLaunchKernelGGL(affw_init, dim3((B_ * Q_ + 255) / 256), dim3(256), 0, stream, gcnt);
  hipLaunchKernelGGL(affw_main, dim3(NWG_, B_), dim3(256), 0, stream,
                     key, query, mask, wv, awv, out, gpart, gcnt);
}
